// Encoder_36610301231730
// MI455X (gfx1250) — matrix operand order repaired and checked by reading, not compiled, not run on hardware
//
#include <hip/hip_runtime.h>
#include <hip/hip_bf16.h>

typedef __attribute__((ext_vector_type(16))) _Float16 v16h;
typedef __attribute__((ext_vector_type(8)))  float    v8f;

#define T_LEN 1000
#define CC    512
#define HH    256
#define AS_STRIDE 520   // LDS row stride in halves (512 + 8 pad -> conflict-free)

union AFrag { v16h v; float4 f4[2]; };   // 32 bytes: two B128 chunks

__device__ __forceinline__ float sigm(float x) { return 1.0f / (1.0f + __expf(-x)); }

static __device__ __forceinline__ v8f wmma16(const v16h& a, const v16h& b, const v8f& c) {
    return __builtin_amdgcn_wmma_f32_16x16x32_f16(false, a, false, b, (short)0, c, false, false);
}

// A-operand: lane holds two 16B runs at +0 and +16 halves (K = kc*32+8*half ..)
static __device__ __forceinline__ void load_a(AFrag& a, const _Float16* p) {
    a.f4[0] = *(const float4*)(p);
    a.f4[1] = *(const float4*)(p + 16);
}
// B-operand: lane holds two 16B runs at +0 and +16 halves (K = kc*32+8*half ..)
static __device__ __forceinline__ void load_b(AFrag& b, const _Float16* p) {
    b.f4[0] = *(const float4*)(p);
    b.f4[1] = *(const float4*)(p + 16);
}

// CDNA5 async global->LDS copy, 16B per lane (tracked by ASYNCcnt)
static __device__ __forceinline__ void async_copy_b128(const _Float16* gsrc, _Float16* ldst) {
    const unsigned long long ga = (unsigned long long)(uintptr_t)gsrc;
    const unsigned int       la = (unsigned int)(uintptr_t)ldst;  // LDS aperture: addr[31:0]
    asm volatile("global_load_async_to_lds_b128 %0, %1, off"
                 :: "v"(la), "v"(ga) : "memory");
}
static __device__ __forceinline__ void wait_async0() {
    asm volatile("s_wait_asynccnt 0" ::: "memory");
}

// ---------------------------------------------------------------------------
// init: fold BN into per-channel scale/bias, zero h state + grid barrier ctr
// ---------------------------------------------------------------------------
__global__ void __launch_bounds__(512) init_kernel(
    const float* __restrict__ cb0, const float* __restrict__ bg0, const float* __restrict__ bb0,
    const float* __restrict__ bm0, const float* __restrict__ bv0,
    const float* __restrict__ cb1, const float* __restrict__ bg1, const float* __restrict__ bb1,
    const float* __restrict__ bm1, const float* __restrict__ bv1,
    const float* __restrict__ cb2, const float* __restrict__ bg2, const float* __restrict__ bb2,
    const float* __restrict__ bm2, const float* __restrict__ bv2,
    float* __restrict__ sb, _Float16* __restrict__ hbuf, int* __restrict__ ctr)
{
    const int l = blockIdx.x;
    const int c = threadIdx.x;
    if (l < 3) {
        const float* cb = l == 0 ? cb0 : (l == 1 ? cb1 : cb2);
        const float* bg = l == 0 ? bg0 : (l == 1 ? bg1 : bg2);
        const float* bb = l == 0 ? bb0 : (l == 1 ? bb1 : bb2);
        const float* bm = l == 0 ? bm0 : (l == 1 ? bm1 : bm2);
        const float* bv = l == 0 ? bv0 : (l == 1 ? bv1 : bv2);
        float s = bg[c] * __frsqrt_rn(bv[c] + 1e-5f);
        sb[l * 1024 + c]       = s;
        sb[l * 1024 + 512 + c] = s * (cb[c] - bm[c]) + bb[c];
    } else {
        for (int i = c; i < 2 * 2 * 32 * 256; i += 512) hbuf[i] = (_Float16)0.0f;
        if (c == 0) *ctr = 0;
    }
}

// one-time f32 -> f16 cast (wih / whh)
__global__ void __launch_bounds__(256) cast_f32_f16(
    const float* __restrict__ s, _Float16* __restrict__ d, int n)
{
    int i = blockIdx.x * 256 + threadIdx.x;
    if (i < n) d[i] = (_Float16)s[i];
}

// one-time conv weight repack: Wt[n][k] f16, k = dk*cin + ci, zero pad to kpad
__global__ void __launch_bounds__(256) repack_conv_w(
    const float* __restrict__ W, _Float16* __restrict__ wt, int cin, int ktot, int kpad)
{
    const int n = blockIdx.x;
    for (int k = threadIdx.x; k < kpad; k += 256) {
        _Float16 v = (_Float16)0.f;
        if (k < ktot) {
            const int dk = k / cin, ci = k - dk * cin;
            v = (_Float16)W[((size_t)n * cin + ci) * 5 + dk];
        }
        wt[(size_t)n * kpad + k] = v;
    }
}

// ---------------------------------------------------------------------------
// Layer-0 conv (Cin=80, K=400): small-K implicit GEMM, scalar gather A path.
// grid=(63 time-tiles, 32 batch); wave w owns N-tiles {4w..4w+3}.
// ---------------------------------------------------------------------------
__global__ void __launch_bounds__(256) conv0_bn_relu(
    const float*    __restrict__ x0,   // (B,80,T) f32
    const _Float16* __restrict__ wt,   // repacked (512, 416) f16
    const float*    __restrict__ sc, const float* __restrict__ bs,
    _Float16*       __restrict__ y)    // (B,T,512) f16
{
    constexpr int CIN = 80, KTOT = 400, NKC = 13, KPAD = 416;
    const int mt   = blockIdx.x;
    const int b    = blockIdx.y;
    const int wave = threadIdx.x >> 5;
    const int lane = threadIdx.x & 31;
    const int l16  = lane & 15;
    const int half = lane >> 4;
    const int t0   = mt * 16;

    auto loadA = [&](AFrag& a, int kc) {
#pragma unroll
        for (int i = 0; i < 8; ++i) {
            const int kb = (i < 4 ? 2 * i : 16 + 2 * (i - 4)) + 8 * half;
#pragma unroll
            for (int j = 0; j < 2; ++j) {
                const int k = kc * 32 + kb + j;
                _Float16 v = (_Float16)0.f;
                if (k < KTOT) {
                    const int dk = k / CIN;
                    const int ci = k - dk * CIN;
                    const int t  = t0 + l16 + dk - 2;
                    if (t >= 0 && t < T_LEN)
                        v = (_Float16)x0[((size_t)b * CIN + ci) * T_LEN + t];
                }
                a.v[2 * i + j] = v;
            }
        }
    };
    auto loadB4 = [&](AFrag* bu, int kc) {
#pragma unroll
        for (int q = 0; q < 4; ++q) {
            const int n = (wave * 4 + q) * 16 + l16;
            load_b(bu[q], wt + (size_t)n * KPAD + kc * 32 + half * 8);
        }
    };

    const v8f Z = (v8f){0.f, 0.f, 0.f, 0.f, 0.f, 0.f, 0.f, 0.f};
    v8f acc[4];
#pragma unroll
    for (int q = 0; q < 4; ++q) acc[q] = Z;

    AFrag aX, aY;
    AFrag bX[4], bY[4];
    loadA(aX, 0);
    loadB4(bX, 0);
    for (int kc = 0; kc < NKC; kc += 2) {
        if (kc + 1 < NKC) { loadA(aY, kc + 1); loadB4(bY, kc + 1); }
#pragma unroll
        for (int q = 0; q < 4; ++q) acc[q] = wmma16(aX.v, bX[q].v, acc[q]);
        if (kc + 2 < NKC) { loadA(aX, kc + 2); loadB4(bX, kc + 2); }
        if (kc + 1 < NKC) {
#pragma unroll
            for (int q = 0; q < 4; ++q) acc[q] = wmma16(aY.v, bY[q].v, acc[q]);
        }
    }
#pragma unroll
    for (int q = 0; q < 4; ++q) {
        const int n = (wave * 4 + q) * 16 + l16;
        const float s = sc[n], o = bs[n];
#pragma unroll
        for (int r = 0; r < 8; ++r) {
            const int t = t0 + r + 8 * half;
            if (t < T_LEN) {
                float v = s * acc[q][r] + o;
                v = v > 0.f ? v : 0.f;
                y[((size_t)b * T_LEN + t) * CC + n] = (_Float16)v;
            }
        }
    }
}

// ---------------------------------------------------------------------------
// Layers 1/2 conv (Cin=512): tall-M implicit GEMM exploiting 320KB LDS.
// Block = 128 time rows x 128 cout. A tile (132 halo rows x 512 f16, 137KB)
// async-staged once, halo zero-filled. Wave owns 1 N-tile x 8 M-tiles, so each
// streamed B fragment feeds 8 WMMAs (B L2 traffic ~0.65 GB/layer).
// grid=(8 m-blocks, 4 n-blocks, 32 batch).
// ---------------------------------------------------------------------------
__global__ void __launch_bounds__(256) conv_big_bn_relu(
    const _Float16* __restrict__ xin,  // (B,T,512) f16
    const _Float16* __restrict__ wt,   // repacked (512, 2560) f16
    const float*    __restrict__ sc, const float* __restrict__ bs,
    _Float16*       __restrict__ y)    // (B,T,512) f16
{
    constexpr int NKC = 80, KPAD = 2560, ROWS = 132;
    __shared__ _Float16 a_s[ROWS * AS_STRIDE];   // 137,280 B

    const int mb   = blockIdx.x;       // 0..7  (128 t rows each)
    const int nb   = blockIdx.y;       // 0..3  (128 cout each)
    const int b    = blockIdx.z;       // batch
    const int wave = threadIdx.x >> 5;
    const int lane = threadIdx.x & 31;
    const int l16  = lane & 15;
    const int half = lane >> 4;
    const int t0   = mb * 128;
    const int n    = (nb * 8 + wave) * 16 + l16;   // this wave's output column

    // stage A tile: rows r=0..131 <-> t = t0+r-2; 64 x 16B chunks per row
    for (int c = threadIdx.x; c < ROWS * 64; c += 256) {
        const int row = c >> 6;
        const int col = (c & 63) * 8;
        _Float16* ldst = a_s + row * AS_STRIDE + col;
        const int t = t0 + row - 2;
        if (t >= 0 && t < T_LEN) {
            async_copy_b128(xin + ((size_t)b * T_LEN + t) * CC + col, ldst);
        } else {
            *(float4*)ldst = (float4){0.f, 0.f, 0.f, 0.f};
        }
    }
    wait_async0();
    __syncthreads();

    auto loadA = [&](AFrag& a, int mt, int kc) {
#pragma unroll
        for (int c = 0; c < 2; ++c) {
            const int kb = kc * 32 + 8 * half + 16 * c;
            const int dk = kb >> 9;            // /512
            const int ci = kb & 511;
            a.f4[c] = *(const float4*)(a_s + (mt * 16 + l16 + dk) * AS_STRIDE + ci);
        }
    };
    const _Float16* wrow = wt + (size_t)n * KPAD + half * 8;

    const v8f Z = (v8f){0.f, 0.f, 0.f, 0.f, 0.f, 0.f, 0.f, 0.f};
    v8f acc[8];
#pragma unroll
    for (int mt = 0; mt < 8; ++mt) acc[mt] = Z;

    AFrag bX, bY, a;
    load_b(bX, wrow);
    for (int kc = 0; kc < NKC; kc += 2) {
        load_b(bY, wrow + (kc + 1) * 32);
#pragma unroll
        for (int mt = 0; mt < 8; ++mt) {
            loadA(a, mt, kc);
            acc[mt] = wmma16(a.v, bX.v, acc[mt]);
        }
        if (kc + 2 < NKC) load_b(bX, wrow + (kc + 2) * 32);
#pragma unroll
        for (int mt = 0; mt < 8; ++mt) {
            loadA(a, mt, kc + 1);
            acc[mt] = wmma16(a.v, bY.v, acc[mt]);
        }
    }

    const float s = sc[n], o = bs[n];
#pragma unroll
    for (int mt = 0; mt < 8; ++mt) {
#pragma unroll
        for (int r = 0; r < 8; ++r) {
            const int t = t0 + mt * 16 + r + 8 * half;
            if (t < T_LEN) {
                float v = s * acc[mt][r] + o;
                v = v > 0.f ? v : 0.f;
                y[((size_t)b * T_LEN + t) * CC + n] = (_Float16)v;
            }
        }
    }
}

// ---------------------------------------------------------------------------
// LSTM input projection, tall-M: gates[d][t][b][n] = x3[b][t][:].wih_d[n][:]
// Block = 128 (t,b) rows (4 t x 32 b) x 128 gate cols; A (128x512, 133KB LDS)
// async-staged; wave owns 1 N-tile x 8 M-tiles (wih traffic ~0.5 GB total).
// grid=(250 m-blocks, 8 n-blocks, 2 dirs).
// ---------------------------------------------------------------------------
__global__ void __launch_bounds__(256) lstm_input_gates(
    const _Float16* __restrict__ x3,
    const _Float16* __restrict__ wih_h,   // [2][1024][512] f16
    _Float16* __restrict__ gates)         // [2][1000][32][1024] f16
{
    __shared__ _Float16 a_s[128 * AS_STRIDE];   // 133,120 B

    const int mb = blockIdx.x;            // 0..249 -> t in [4mb, 4mb+4)
    const int nb = blockIdx.y;            // 0..7
    const int d  = blockIdx.z;
    const _Float16* wih = wih_h + (size_t)d * 1024 * CC;
    const int wave = threadIdx.x >> 5;
    const int lane = threadIdx.x & 31;
    const int l16  = lane & 15;
    const int half = lane >> 4;
    const int n    = (nb * 8 + wave) * 16 + l16;

    // stage A tile: local row r -> t = 4mb + (r>>5), batch = r&31
    for (int c = threadIdx.x; c < 128 * 64; c += 256) {
        const int row = c >> 6;
        const int col = (c & 63) * 8;
        const int t   = mb * 4 + (row >> 5);
        const int bb  = row & 31;
        async_copy_b128(x3 + (((size_t)bb * T_LEN + t)) * CC + col,
                        a_s + row * AS_STRIDE + col);
    }
    wait_async0();
    __syncthreads();

    const _Float16* wrow = wih + (size_t)n * CC + half * 8;

    const v8f Z = (v8f){0.f, 0.f, 0.f, 0.f, 0.f, 0.f, 0.f, 0.f};
    v8f acc[8];
#pragma unroll
    for (int mt = 0; mt < 8; ++mt) acc[mt] = Z;

    AFrag bX, bY, a;
    load_b(bX, wrow);
#pragma unroll
    for (int kc = 0; kc < 16; kc += 2) {
        load_b(bY, wrow + (kc + 1) * 32);
#pragma unroll
        for (int mt = 0; mt < 8; ++mt) {
            load_a(a, a_s + (mt * 16 + l16) * AS_STRIDE + kc * 32 + 8 * half);
            acc[mt] = wmma16(a.v, bX.v, acc[mt]);
        }
        if (kc + 2 < 16) load_b(bX, wrow + (kc + 2) * 32);
#pragma unroll
        for (int mt = 0; mt < 8; ++mt) {
            load_a(a, a_s + (mt * 16 + l16) * AS_STRIDE + (kc + 1) * 32 + 8 * half);
            acc[mt] = wmma16(a.v, bY.v, acc[mt]);
        }
    }

#pragma unroll
    for (int mt = 0; mt < 8; ++mt) {
#pragma unroll
        for (int r = 0; r < 8; ++r) {
            const int gr = mt * 16 + r + 8 * half;     // local row 0..127
            const int t  = mb * 4 + (gr >> 5);
            const int bb = gr & 31;
            gates[(((size_t)d * T_LEN + t) * 32 + bb) * 1024 + n] = (_Float16)acc[mt][r];
        }
    }
}

// ---------------------------------------------------------------------------
// Persistent bidirectional LSTM recurrence with zoneout. 16 WGs = 2 dirs x 8.
// WG owns h cols [32wg,32wg+32) and gate rows {g*256+32wg+jj}; whh B-fragments
// register-resident; A fragments double-buffered; gate staging + c in LDS;
// h double-buffered in global (L2-hot); device-scope s_sleep spin barrier.
// ---------------------------------------------------------------------------
__global__ void __launch_bounds__(256) bilstm_recurrent(
    const _Float16* __restrict__ gates_x,  // [2][1000][32][1024]
    const _Float16* __restrict__ whh_h,    // [2][1024][256] f16
    const float* __restrict__ bih_f, const float* __restrict__ bhh_f,
    const float* __restrict__ bih_b, const float* __restrict__ bhh_b,
    _Float16* __restrict__ hbuf,           // [2 dirs][2 parity][32][256]
    float* __restrict__ out,               // [32][1000][512]
    int* __restrict__ barrier_ctr)
{
    __shared__ float gates_s[32 * 128];
    __shared__ float c_s[32 * 32];

    const int blk  = blockIdx.x;   // 0..15
    const int d    = blk >> 3;
    const int wg   = blk & 7;
    const int wave = threadIdx.x >> 5;
    const int lane = threadIdx.x & 31;
    const int l16  = lane & 15;
    const int half = lane >> 4;

    const _Float16* whh = whh_h + (size_t)d * 1024 * HH;
    const float* bih = d ? bih_b : bih_f;
    const float* bhh = d ? bhh_b : bhh_f;

    const int lc   = 16 * wave + l16;                  // local col in [0,128)
    const int grow = (lc >> 5) * 256 + 32 * wg + (lc & 31);
    v16h bfrag[8];
#pragma unroll
    for (int kc = 0; kc < 8; ++kc) {
        AFrag bu;
        load_b(bu, whh + (size_t)grow * HH + kc * 32 + half * 8);
        bfrag[kc] = bu.v;
    }

    for (int e = threadIdx.x; e < 32 * 32; e += 256) c_s[e] = 0.f;
    __syncthreads();

    volatile int* vctr = barrier_ctr;

    for (int s = 0; s < T_LEN; ++s) {
        const int p = s & 1;
        const _Float16* hin  = hbuf + ((size_t)(d * 2 + p)) * 32 * HH;
        _Float16*       hout = hbuf + ((size_t)(d * 2 + (1 - p))) * 32 * HH;

        const int t_x = d ? (T_LEN - 1 - s) : s;
        const _Float16* gx = gates_x + (((size_t)d * T_LEN + t_x) * 32) * 1024;
        {
            const int tid = threadIdx.x;
            const _Float16* pa = gx + (tid >> 3) * 1024 + (((tid & 7) >> 1) * 256)
                                    + 32 * wg + (tid & 1) * 16;
            __builtin_prefetch(pa, 0, 1);
        }

        const v8f Z = (v8f){0.f, 0.f, 0.f, 0.f, 0.f, 0.f, 0.f, 0.f};
        v8f acc0 = Z, acc1 = Z;
        const _Float16* hrow0 = hin + (size_t)l16 * HH + 8 * half;
        const _Float16* hrow1 = hin + (size_t)(16 + l16) * HH + 8 * half;

        AFrag a0[2], a1[2];
        load_a(a0[0], hrow0);
        load_a(a1[0], hrow1);
#pragma unroll
        for (int kc = 0; kc < 8; ++kc) {
            if (kc < 7) {
                load_a(a0[(kc + 1) & 1], hrow0 + (kc + 1) * 32);
                load_a(a1[(kc + 1) & 1], hrow1 + (kc + 1) * 32);
            }
            acc0 = wmma16(a0[kc & 1].v, bfrag[kc], acc0);
            acc1 = wmma16(a1[kc & 1].v, bfrag[kc], acc1);
        }
#pragma unroll
        for (int r = 0; r < 8; ++r) {
            gates_s[(r + 8 * half) * 128 + lc]      = acc0[r];
            gates_s[(16 + r + 8 * half) * 128 + lc] = acc1[r];
        }
        __syncthreads();

#pragma unroll
        for (int e4 = 0; e4 < 4; ++e4) {
            const int e  = threadIdx.x + 256 * e4;   // 1024 = 32 batch x 32 cols
            const int b  = e >> 5;
            const int jj = e & 31;
            const int hcol = 32 * wg + jj;
            const size_t gxb = (size_t)b * 1024;

            float gi = gates_s[b * 128 +      jj] + (float)gx[gxb +       hcol] + bih[      hcol] + bhh[      hcol];
            float gf = gates_s[b * 128 + 32 + jj] + (float)gx[gxb + 256 + hcol] + bih[256 + hcol] + bhh[256 + hcol];
            float gg = gates_s[b * 128 + 64 + jj] + (float)gx[gxb + 512 + hcol] + bih[512 + hcol] + bhh[512 + hcol];
            float go = gates_s[b * 128 + 96 + jj] + (float)gx[gxb + 768 + hcol] + bih[768 + hcol] + bhh[768 + hcol];

            const float c_old = c_s[b * 32 + jj];
            const float c2 = sigm(gf) * c_old + sigm(gi) * tanhf(gg);
            const float h2 = sigm(go) * tanhf(c2);
            const float h_old = (float)hin[b * HH + hcol];
            const float cn = 0.1f * c_old + 0.9f * c2;     // zoneout (deterministic)
            const float hn = 0.1f * h_old + 0.9f * h2;

            c_s[b * 32 + jj] = cn;
            hout[b * HH + hcol] = (_Float16)hn;
            out[((size_t)b * T_LEN + t_x) * (2 * HH) + d * HH + hcol] = hn;
        }

        __threadfence();
        __syncthreads();
        if (threadIdx.x == 0) {
            atomicAdd(barrier_ctr, 1);
            while (*vctr < 16 * (s + 1)) {
                __builtin_amdgcn_s_sleep(1);
            }
        }
        __syncthreads();
        __threadfence();
    }
}

// ---------------------------------------------------------------------------
// launch
// ---------------------------------------------------------------------------
extern "C" void kernel_launch(void* const* d_in, const int* in_sizes, int n_in,
                              void* d_out, int out_size, void* d_ws, size_t ws_size,
                              hipStream_t stream) {
    const float* x     = (const float*)d_in[0];
    const float* cw0   = (const float*)d_in[1];
    const float* cb0   = (const float*)d_in[2];
    const float* bg0   = (const float*)d_in[3];
    const float* bb0   = (const float*)d_in[4];
    const float* bm0   = (const float*)d_in[5];
    const float* bv0   = (const float*)d_in[6];
    const float* cw1   = (const float*)d_in[7];
    const float* cb1   = (const float*)d_in[8];
    const float* bg1   = (const float*)d_in[9];
    const float* bb1   = (const float*)d_in[10];
    const float* bm1   = (const float*)d_in[11];
    const float* bv1   = (const float*)d_in[12];
    const float* cw2   = (const float*)d_in[13];
    const float* cb2   = (const float*)d_in[14];
    const float* bg2   = (const float*)d_in[15];
    const float* bb2   = (const float*)d_in[16];
    const float* bm2   = (const float*)d_in[17];
    const float* bv2   = (const float*)d_in[18];
    const float* wih_f = (const float*)d_in[19];
    const float* whh_f = (const float*)d_in[20];
    const float* bih_f = (const float*)d_in[21];
    const float* bhh_f = (const float*)d_in[22];
    const float* wih_b = (const float*)d_in[23];
    const float* whh_b = (const float*)d_in[24];
    const float* bih_b = (const float*)d_in[25];
    const float* bhh_b = (const float*)d_in[26];

    char* ws = (char*)d_ws;
    const size_t XB = (size_t)32 * T_LEN * CC * sizeof(_Float16);            // 32.77 MB
    const size_t GB = (size_t)2 * T_LEN * 32 * 1024 * sizeof(_Float16);      // 131 MB
    size_t o = 0;
    _Float16* xb0   = (_Float16*)(ws + o); o += XB;
    _Float16* xb1   = (_Float16*)(ws + o); o += XB;
    _Float16* gptr  = (_Float16*)(ws + o); o += GB;
    _Float16* hbuf  = (_Float16*)(ws + o); o += 65536;
    float*    sb    = (float*)   (ws + o); o += 16384;
    int*      ctr   = (int*)     (ws + o); o += 256;
    _Float16* w0t   = (_Float16*)(ws + o); o += (size_t)512 * 416  * 2;      // kpad 416
    _Float16* w1t   = (_Float16*)(ws + o); o += (size_t)512 * 2560 * 2;
    _Float16* w2t   = (_Float16*)(ws + o); o += (size_t)512 * 2560 * 2;
    _Float16* wihh  = (_Float16*)(ws + o); o += (size_t)2 * 1024 * 512 * 2;  // f then b
    _Float16* whhh  = (_Float16*)(ws + o); o += (size_t)2 * 1024 * 256 * 2;

    init_kernel<<<dim3(4), 512, 0, stream>>>(cb0, bg0, bb0, bm0, bv0,
                                             cb1, bg1, bb1, bm1, bv1,
                                             cb2, bg2, bb2, bm2, bv2,
                                             sb, hbuf, ctr);

    // one-time weight repacks / casts
    repack_conv_w<<<dim3(512), 256, 0, stream>>>(cw0, w0t,  80,  400,  416);
    repack_conv_w<<<dim3(512), 256, 0, stream>>>(cw1, w1t, 512, 2560, 2560);
    repack_conv_w<<<dim3(512), 256, 0, stream>>>(cw2, w2t, 512, 2560, 2560);
    const int NWIH = 1024 * 512, NWHH = 1024 * 256;
    cast_f32_f16<<<dim3((NWIH + 255) / 256), 256, 0, stream>>>(wih_f, wihh,        NWIH);
    cast_f32_f16<<<dim3((NWIH + 255) / 256), 256, 0, stream>>>(wih_b, wihh + NWIH, NWIH);
    cast_f32_f16<<<dim3((NWHH + 255) / 256), 256, 0, stream>>>(whh_f, whhh,        NWHH);
    cast_f32_f16<<<dim3((NWHH + 255) / 256), 256, 0, stream>>>(whh_b, whhh + NWHH, NWHH);

    conv0_bn_relu<<<dim3(63, 32), 256, 0, stream>>>(x, w0t, sb + 0, sb + 512, xb0);
    conv_big_bn_relu<<<dim3(8, 4, 32), 256, 0, stream>>>(xb0, w1t, sb + 1024, sb + 1536, xb1);
    conv_big_bn_relu<<<dim3(8, 4, 32), 256, 0, stream>>>(xb1, w2t, sb + 2048, sb + 2560, xb0);

    lstm_input_gates<<<dim3(250, 8, 2), 256, 0, stream>>>(xb0, wihh, gptr);

    bilstm_recurrent<<<dim3(16), 256, 0, stream>>>(gptr, whhh,
                                                   bih_f, bhh_f, bih_b, bhh_b,
                                                   hbuf, (float*)d_out, ctr);
}
